// GroupedQueryAttention_50568944943561
// MI455X (gfx1250) — matrix operand order repaired and checked by reading, not compiled, not run on hardware
//
#include <hip/hip_runtime.h>

// ---------------------------------------------------------------------------
// GQA attention layer for MI455X (gfx1250, wave32, WMMA).
// All matmuls on v_wmma_f32_16x16x32_bf16 (f32 accumulate), softmax/norm in f32.
// GEMMs use double-buffered LDS tiles staged with GLOBAL_LOAD_ASYNC_TO_LDS_B128
// (ASYNCcnt-tracked) when the builtin is available.
// ---------------------------------------------------------------------------

#define SEQ   2048
#define DIM   2048
#define NH    32
#define NKV   8
#define HD    64
#define KVDIM (NKV * HD)   // 512
#define NEGM  (-1e9f)

typedef __attribute__((ext_vector_type(16))) __bf16 bf16x16;
typedef __attribute__((ext_vector_type(8)))  __bf16 bf16x8;
typedef __attribute__((ext_vector_type(8)))  float  f32x8;

union AFrag { bf16x16 v; bf16x8 h[2]; };

static __device__ inline f32x8 wmma_bf16(bf16x16 a, bf16x16 b, f32x8 c) {
    // 8 args: (neg_a, A, neg_b, B, c_mod, C, reuse_a, reuse_b)
    return __builtin_amdgcn_wmma_f32_16x16x32_bf16(false, a, false, b,
                                                   (short)0, c, false, false);
}

// --------------------- async global -> LDS staging -------------------------

#if __has_builtin(__builtin_amdgcn_global_load_async_to_lds_b128)
#define HAS_ASYNC_LDS 1
typedef __attribute__((__vector_size__(4 * sizeof(int)))) int v4i_t;
static __device__ inline void copy16(const __bf16* g, __bf16* l) {
    __builtin_amdgcn_global_load_async_to_lds_b128((v4i_t*)g, (v4i_t*)l, 0, 0);
}
static __device__ inline void wait_async() {
    asm volatile("s_wait_asynccnt 0x0" ::: "memory");
}
#else
#define HAS_ASYNC_LDS 0
static __device__ inline void copy16(const __bf16* g, __bf16* l) {
    *reinterpret_cast<bf16x8*>(l) = *reinterpret_cast<const bf16x8*>(g);
}
static __device__ inline void wait_async() {}
#endif

// ------------------------------ converters ---------------------------------

__global__ void cvt_bf16(const float* __restrict__ in, __bf16* __restrict__ out, int n) {
    int i = blockIdx.x * blockDim.x + threadIdx.x;
    if (i < n) out[i] = (__bf16)in[i];
}

// W (K x N, row-major f32) -> WT (N x K, row-major bf16)
__global__ void transpose_cvt(const float* __restrict__ in, __bf16* __restrict__ out,
                              int K, int N) {
    int i = blockIdx.x * blockDim.x + threadIdx.x;
    if (i < K * N) {
        int k = i / N, n = i % N;
        out[(size_t)n * K + k] = (__bf16)in[i];
    }
}

// ------------------------------ WMMA GEMM ----------------------------------
// C[M x N] f32 = A[M x K] bf16 (row-major) * BT[N x K] bf16 (row-major = B^T).
// 256-thread block computes a 64x128 tile: 8 waves, each 16x64 (4 accumulators).
// K-loop stages 64x32 A-tile + 128x32 B-tile in double-buffered LDS via the
// async global->LDS path; one barrier per K-step.
#define BM 64
#define BN 128
#define BK 32

__global__ __launch_bounds__(256)
void gemm_bf16(const __bf16* __restrict__ A, const __bf16* __restrict__ BT,
               float* __restrict__ C, int M, int N, int K) {
    __shared__ __align__(32) __bf16 smA[2][BM * BK];   // 2 x 4 KB
    __shared__ __align__(32) __bf16 smB[2][BN * BK];   // 2 x 8 KB

    const int tid  = threadIdx.x;
    const int wave = tid >> 5;
    const int lane = tid & 31;
    const int l15  = lane & 15;
    const int half = lane >> 4;
    const int mw   = (wave & 3) * 16;   // wave M offset inside tile
    const int nw   = (wave >> 2) * 64;  // wave N offset inside tile

    const int bm = blockIdx.y * BM;
    const int bn = blockIdx.x * BN;

    // staging assignments: 16B chunks (8 bf16)
    const int ar  = tid >> 2,           ac  = (tid & 3) * 8;            // A: 256 chunks
    const int br0 = tid >> 2,           bc0 = (tid & 3) * 8;            // B: 512 chunks
    const int br1 = (tid + 256) >> 2,   bc1 = ((tid + 256) & 3) * 8;

    auto stage = [&](int buf, int kc) {
        copy16(A  + (size_t)(bm + ar)  * K + kc + ac,  &smA[buf][ar  * BK + ac]);
        copy16(BT + (size_t)(bn + br0) * K + kc + bc0, &smB[buf][br0 * BK + bc0]);
        copy16(BT + (size_t)(bn + br1) * K + kc + bc1, &smB[buf][br1 * BK + bc1]);
    };

    f32x8 acc0 = {}, acc1 = {}, acc2 = {}, acc3 = {};
    const int nk = K / BK;

    stage(0, 0);
    for (int k = 0; k < nk; ++k) {
        wait_async();           // this wave's staged tile has landed in LDS
        __syncthreads();        // everyone's tile visible; prev-iter reads done
        if (k + 1 < nk) stage((k + 1) & 1, (k + 1) * BK);

        const int buf = k & 1;
        // A 16x32 frag: lane half selects K 0..15 / 16..31
        AFrag a;
        const __bf16* ap = &smA[buf][(mw + l15) * BK + half * 16];
        a.h[0] = *reinterpret_cast<const bf16x8*>(ap);
        a.h[1] = *reinterpret_cast<const bf16x8*>(ap + 8);
        // B 32x16 frags: lane = N column, lane half selects K 0..15 / 16..31
        const __bf16* bp = &smB[buf][(nw + l15) * BK + half * 16];
        bf16x16 b0 = *reinterpret_cast<const bf16x16*>(bp);
        bf16x16 b1 = *reinterpret_cast<const bf16x16*>(bp + 16 * BK);
        bf16x16 b2 = *reinterpret_cast<const bf16x16*>(bp + 32 * BK);
        bf16x16 b3 = *reinterpret_cast<const bf16x16*>(bp + 48 * BK);

        acc0 = wmma_bf16(a.v, b0, acc0);
        acc1 = wmma_bf16(a.v, b1, acc1);
        acc2 = wmma_bf16(a.v, b2, acc2);
        acc3 = wmma_bf16(a.v, b3, acc3);
    }

    // C/D layout: M = r + 8*half, N = l15 (+16*tile)
    #pragma unroll
    for (int r = 0; r < 8; ++r) {
        float* crow = C + (size_t)(bm + mw + r + half * 8) * N + bn + nw + l15;
        crow[0]  = acc0[r];
        crow[16] = acc1[r];
        crow[32] = acc2[r];
        crow[48] = acc3[r];
    }
}

// -------------------- QK RMSNorm + RoPE (rotate-half) ----------------------
// in: (SEQ, nheads*HD) f32 from GEMM; out: head-major (h, s, d) bf16.
__global__ void qknorm_rope(const float* __restrict__ in, const float* __restrict__ w,
                            __bf16* __restrict__ out, const int* __restrict__ start_pos,
                            int in_stride) {
    const int s = blockIdx.x, h = blockIdx.y, t = threadIdx.x;  // t in [0,32)
    const float* row = in + (size_t)s * in_stride + h * HD;
    float x0 = row[t], x1 = row[t + 32];

    float ss = x0 * x0 + x1 * x1;               // sum of squares over 64 elems
    ss += __shfl_xor(ss, 1, 32);
    ss += __shfl_xor(ss, 2, 32);
    ss += __shfl_xor(ss, 4, 32);
    ss += __shfl_xor(ss, 8, 32);
    ss += __shfl_xor(ss, 16, 32);
    const float rn = rsqrtf(ss * (1.0f / HD) + 1e-6f);
    x0 *= rn * w[t];
    x1 *= rn * w[t + 32];

    const float pos  = (float)(start_pos[0] + s);
    const float freq = __expf(-logf(10000.0f) * (float)(2 * t) / (float)HD);
    const float ang  = pos * freq;
    const float cs = __cosf(ang), sn = __sinf(ang);
    const float o0 = x0 * cs - x1 * sn;
    const float o1 = x1 * cs + x0 * sn;

    __bf16* orow = out + ((size_t)h * SEQ + s) * HD;
    orow[t]      = (__bf16)o0;
    orow[t + 32] = (__bf16)o1;
}

// V: (SEQ, KVDIM) f32 -> transposed bf16 (h, d, s) so PV B-frags load contiguously.
__global__ void v_transpose_cvt(const float* __restrict__ in, __bf16* __restrict__ out) {
    const int s = blockIdx.x, h = blockIdx.y, t = threadIdx.x;
    const float* row = in + (size_t)s * KVDIM + h * HD;
    out[((size_t)h * HD + t)      * SEQ + s] = (__bf16)row[t];
    out[((size_t)h * HD + t + 32) * SEQ + s] = (__bf16)row[t + 32];
}

// ------------------------- flash attention (wave) --------------------------
// One wave per (head, 16-query tile). Online softmax over 32-key chunks.
__global__ void attn_kernel(const __bf16* __restrict__ Q, const __bf16* __restrict__ Km,
                            const __bf16* __restrict__ VT, __bf16* __restrict__ O) {
    __shared__ __align__(32) __bf16 Pl[16 * 32];

    const int lane = threadIdx.x;
    const int l15  = lane & 15;
    const int half = lane >> 4;
    const int h    = blockIdx.x;
    const int qt   = blockIdx.y;
    const int kvh  = h >> 2;        // N_REP = 4

    const __bf16* qh = Q  + ((size_t)h   * SEQ + qt * 16) * HD;
    const __bf16* kh = Km + (size_t)kvh * SEQ * HD;
    const __bf16* vh = VT + (size_t)kvh * HD  * SEQ;

    // Q fragments for d=0..31 and d=32..63
    AFrag aq0, aq1;
    {
        const __bf16* ap = qh + (size_t)l15 * HD + half * 16;
        aq0.h[0] = *reinterpret_cast<const bf16x8*>(ap);
        aq0.h[1] = *reinterpret_cast<const bf16x8*>(ap + 8);
        aq1.h[0] = *reinterpret_cast<const bf16x8*>(ap + 32);
        aq1.h[1] = *reinterpret_cast<const bf16x8*>(ap + 40);
    }

    f32x8 o0 = {}, o1 = {}, o2 = {}, o3 = {};
    float mrow[8], lrow[8];
    #pragma unroll
    for (int r = 0; r < 8; ++r) { mrow[r] = -1e30f; lrow[r] = 0.0f; }

    const int limit   = (qt + 1) * 16;        // causal: keys < limit
    const int nchunks = (limit + 31) >> 5;

    for (int c = 0; c < nchunks; ++c) {
        const int kb = c * 32;

        // prefetch next chunk's K rows / V columns (global_prefetch_b8)
        if (c + 1 < nchunks) {
            __builtin_prefetch(kh + (size_t)(kb + 32 + lane) * HD, 0, 1);
            __builtin_prefetch(vh + (size_t)lane * SEQ + kb + 32, 0, 1);
        }

        // K^T fragments: B[d][n] = K[key=n][d]; lane = key, half = K-dim chunk
        const __bf16* k0 = kh + (size_t)(kb + l15) * HD + half * 16;
        const __bf16* k1 = kh + (size_t)(kb + 16 + l15) * HD + half * 16;
        bf16x16 bk00 = *reinterpret_cast<const bf16x16*>(k0);
        bf16x16 bk01 = *reinterpret_cast<const bf16x16*>(k0 + 32);
        bf16x16 bk10 = *reinterpret_cast<const bf16x16*>(k1);
        bf16x16 bk11 = *reinterpret_cast<const bf16x16*>(k1 + 32);

        f32x8 s0 = {}, s1 = {};
        s0 = wmma_bf16(aq0.v, bk00, s0);
        s0 = wmma_bf16(aq1.v, bk01, s0);
        s1 = wmma_bf16(aq0.v, bk10, s1);
        s1 = wmma_bf16(aq1.v, bk11, s1);

        const int key0 = kb + l15, key1 = key0 + 16;
        float p0[8], p1[8];
        #pragma unroll
        for (int r = 0; r < 8; ++r) {
            const int qrow = qt * 16 + r + half * 8;
            float v0 = s0[r] * 0.125f + (key0 > qrow ? NEGM : 0.0f);
            float v1 = s1[r] * 0.125f + (key1 > qrow ? NEGM : 0.0f);
            float rmax = fmaxf(v0, v1);                 // row reduce within 16-lane half
            rmax = fmaxf(rmax, __shfl_xor(rmax, 1, 32));
            rmax = fmaxf(rmax, __shfl_xor(rmax, 2, 32));
            rmax = fmaxf(rmax, __shfl_xor(rmax, 4, 32));
            rmax = fmaxf(rmax, __shfl_xor(rmax, 8, 32));
            const float mnew  = fmaxf(mrow[r], rmax);
            const float alpha = __expf(mrow[r] - mnew);
            mrow[r] = mnew;
            const float e0 = __expf(v0 - mnew);
            const float e1 = __expf(v1 - mnew);
            float rs = e0 + e1;
            rs += __shfl_xor(rs, 1, 32);
            rs += __shfl_xor(rs, 2, 32);
            rs += __shfl_xor(rs, 4, 32);
            rs += __shfl_xor(rs, 8, 32);
            lrow[r] = lrow[r] * alpha + rs;
            o0[r] *= alpha; o1[r] *= alpha; o2[r] *= alpha; o3[r] *= alpha;
            p0[r] = e0; p1[r] = e1;
        }

        // P (C-layout) -> LDS -> A-layout fragment for PV
        #pragma unroll
        for (int r = 0; r < 8; ++r) {
            const int Mr = r + half * 8;
            Pl[Mr * 32 + l15]      = (__bf16)p0[r];
            Pl[Mr * 32 + 16 + l15] = (__bf16)p1[r];
        }
        __syncthreads();
        AFrag pa;
        {
            const __bf16* pp = &Pl[l15 * 32 + half * 16];
            pa.h[0] = *reinterpret_cast<const bf16x8*>(pp);
            pa.h[1] = *reinterpret_cast<const bf16x8*>(pp + 8);
        }
        __syncthreads();

        // V fragments: B[k][n] = V[kb+k][hd = t*16+n] = VT[hd][kb+k] (contiguous k)
        const int ksv = kb + half * 16;
        bf16x16 bv0 = *reinterpret_cast<const bf16x16*>(vh + (size_t)( 0 + l15) * SEQ + ksv);
        bf16x16 bv1 = *reinterpret_cast<const bf16x16*>(vh + (size_t)(16 + l15) * SEQ + ksv);
        bf16x16 bv2 = *reinterpret_cast<const bf16x16*>(vh + (size_t)(32 + l15) * SEQ + ksv);
        bf16x16 bv3 = *reinterpret_cast<const bf16x16*>(vh + (size_t)(48 + l15) * SEQ + ksv);

        o0 = wmma_bf16(pa.v, bv0, o0);
        o1 = wmma_bf16(pa.v, bv1, o1);
        o2 = wmma_bf16(pa.v, bv2, o2);
        o3 = wmma_bf16(pa.v, bv3, o3);
    }

    // normalize and write (s, h*64+d) bf16 for the Wo GEMM
    #pragma unroll
    for (int r = 0; r < 8; ++r) {
        const float inv = 1.0f / lrow[r];
        const int row = qt * 16 + r + half * 8;
        __bf16* orow = O + (size_t)row * DIM + h * HD + l15;
        orow[0]  = (__bf16)(o0[r] * inv);
        orow[16] = (__bf16)(o1[r] * inv);
        orow[32] = (__bf16)(o2[r] * inv);
        orow[48] = (__bf16)(o3[r] * inv);
    }
}

// ------------------------------- launcher ----------------------------------

extern "C" void kernel_launch(void* const* d_in, const int* in_sizes, int n_in,
                              void* d_out, int out_size, void* d_ws, size_t ws_size,
                              hipStream_t stream) {
    (void)in_sizes; (void)n_in; (void)out_size; (void)ws_size;

    const float* x    = (const float*)d_in[0];
    const float* Wq   = (const float*)d_in[1];
    const float* Wk   = (const float*)d_in[2];
    const float* Wv   = (const float*)d_in[3];
    const float* Wo   = (const float*)d_in[4];
    const float* qn_w = (const float*)d_in[5];
    const float* kn_w = (const float*)d_in[6];
    // d_in[7] = mask: unused, causal mask applied analytically in-kernel
    const int* start_pos = (const int*)d_in[8];

    char* ws = (char*)d_ws;
    size_t off = 0;
    auto carve = [&](size_t bytes) {
        char* p = ws + off;
        off += (bytes + 255) & ~(size_t)255;
        return p;
    };

    __bf16* xb    = (__bf16*)carve((size_t)SEQ * DIM   * 2);
    __bf16* WqT   = (__bf16*)carve((size_t)DIM * DIM   * 2);
    __bf16* WkT   = (__bf16*)carve((size_t)DIM * KVDIM * 2);
    __bf16* WvT   = (__bf16*)carve((size_t)DIM * KVDIM * 2);
    __bf16* WoT   = (__bf16*)carve((size_t)DIM * DIM   * 2);
    float*  q_f32 = (float*) carve((size_t)SEQ * DIM   * 4);
    float*  k_f32 = (float*) carve((size_t)SEQ * KVDIM * 4);
    float*  v_f32 = (float*) carve((size_t)SEQ * KVDIM * 4);
    __bf16* qb    = (__bf16*)carve((size_t)NH  * SEQ * HD * 2);
    __bf16* kbuf  = (__bf16*)carve((size_t)NKV * SEQ * HD * 2);
    __bf16* vT    = (__bf16*)carve((size_t)NKV * HD  * SEQ * 2);
    __bf16* attnb = (__bf16*)carve((size_t)SEQ * DIM * 2);

    const int T = 256;
    // 1) convert activations + weights to bf16 (weights transposed for B-operand)
    cvt_bf16<<<(SEQ * DIM + T - 1) / T, T, 0, stream>>>(x, xb, SEQ * DIM);
    transpose_cvt<<<(DIM * DIM   + T - 1) / T, T, 0, stream>>>(Wq, WqT, DIM, DIM);
    transpose_cvt<<<(DIM * KVDIM + T - 1) / T, T, 0, stream>>>(Wk, WkT, DIM, KVDIM);
    transpose_cvt<<<(DIM * KVDIM + T - 1) / T, T, 0, stream>>>(Wv, WvT, DIM, KVDIM);
    transpose_cvt<<<(DIM * DIM   + T - 1) / T, T, 0, stream>>>(Wo, WoT, DIM, DIM);

    // 2) QKV projections (WMMA + async-LDS staged tiles)
    gemm_bf16<<<dim3(DIM / BN,   SEQ / BM), 256, 0, stream>>>(xb, WqT, q_f32, SEQ, DIM,   DIM);
    gemm_bf16<<<dim3(KVDIM / BN, SEQ / BM), 256, 0, stream>>>(xb, WkT, k_f32, SEQ, KVDIM, DIM);
    gemm_bf16<<<dim3(KVDIM / BN, SEQ / BM), 256, 0, stream>>>(xb, WvT, v_f32, SEQ, KVDIM, DIM);

    // 3) QK-norm + RoPE -> head-major bf16; V -> (h, d, s) bf16
    qknorm_rope<<<dim3(SEQ, NH),  32, 0, stream>>>(q_f32, qn_w, qb,   start_pos, DIM);
    qknorm_rope<<<dim3(SEQ, NKV), 32, 0, stream>>>(k_f32, kn_w, kbuf, start_pos, KVDIM);
    v_transpose_cvt<<<dim3(SEQ, NKV), 32, 0, stream>>>(v_f32, vT);

    // 4) causal flash attention (WMMA)
    attn_kernel<<<dim3(NH, SEQ / 16), 32, 0, stream>>>(qb, kbuf, vT, attnb);

    // 5) output projection straight into d_out (f32)
    gemm_bf16<<<dim3(DIM / BN, SEQ / BM), 256, 0, stream>>>(attnb, WoT, (float*)d_out,
                                                            SEQ, DIM, DIM);
}
